// ParallelAttnBlock_40896678593073
// MI455X (gfx1250) — compile-verified, matrix operand order checked by reading, not run on hardware
//
#include <hip/hip_runtime.h>

// Problem constants (match reference)
#define BB   2
#define CC   512
#define GG   32
#define CPG  (CC/GG)     // 16 channels per group
#define HWN  4096        // 64*64 tokens
#define EPSV 1e-6f

typedef __attribute__((ext_vector_type(16))) _Float16 v16h;
typedef __attribute__((ext_vector_type(8)))  _Float16 v8h;
typedef __attribute__((ext_vector_type(8)))  float    v8f;

// ---- WMMA helper: D = A(16x32 f16) * B(32x16 f16) + C(f32) ----
__device__ __forceinline__ v8f wmma16(v16h a, v16h b, v8f c) {
  return __builtin_amdgcn_wmma_f32_16x16x32_f16(
      /*neg_a=*/false, a, /*neg_b=*/false, b,
      /*c_mod=*/(short)0, c, /*reuse_a=*/false, /*reuse_b=*/false);
}

// Load a 16-half WMMA fragment: 8 contiguous halves at p, 8 at p+16.
// Matches ISA A/B layout for 16x32 f16 when p = row_base + (lane/16)*8.
__device__ __forceinline__ v16h ldfrag(const _Float16* p) {
  v8h lo = *(const v8h*)p;
  v8h hi = *(const v8h*)(p + 16);
  return __builtin_shufflevector(lo, hi, 0,1,2,3,4,5,6,7,8,9,10,11,12,13,14,15);
}

// ---- CDNA5 async global->LDS copy (ASYNCcnt-tracked), 16B per lane ----
// VDST = 32-bit LDS byte address, ADDR = 64-bit global address, saddr off.
__device__ __forceinline__ void async_b128(const void* g, const _Float16* lds_p) {
  unsigned lds_off = (unsigned)(size_t)lds_p;  // shared ptr low 32 bits = LDS addr
  asm volatile("global_load_async_to_lds_b128 %0, %1, off"
               :: "v"(lds_off), "v"(g) : "memory");
}
#define WAIT_ASYNC_32() asm volatile("s_wait_asynccnt 0x20" ::: "memory")
#define WAIT_ASYNC_0()  asm volatile("s_wait_asynccnt 0x0"  ::: "memory")

// Half-wave (16-lane) reductions; xor masks < 16 never cross the 16-lane halves.
__device__ __forceinline__ float rmax16(float v) {
  for (int m = 8; m >= 1; m >>= 1) v = fmaxf(v, __shfl_xor(v, m, 32));
  return v;
}
__device__ __forceinline__ float rsum16(float v) {
  for (int m = 8; m >= 1; m >>= 1) v += __shfl_xor(v, m, 32);
  return v;
}

// ---------------- Kernel 0: f32 -> f16 weight convert ----------------
__global__ __launch_bounds__(256) void f2h_kernel(const float* __restrict__ src,
                                                  _Float16* __restrict__ dst, int n) {
  int i = blockIdx.x * blockDim.x + threadIdx.x;
  if (i < n) dst[i] = (_Float16)src[i];
}

// ---------------- Kernel 1: GroupNorm -> f16 token-major ----------------
__global__ __launch_bounds__(256) void gn_kernel(const float* __restrict__ x,
                                                 const float* __restrict__ gw,
                                                 const float* __restrict__ gb,
                                                 _Float16* __restrict__ h16) {
  const int bg = blockIdx.x;
  const int b = bg / GG, g = bg % GG;
  const float* xp = x + ((size_t)b * CC + g * CPG) * HWN;
  const int N = CPG * HWN; // 65536
  float s = 0.f, s2 = 0.f;
  for (int i = threadIdx.x; i < N; i += 256) {
    float v = xp[i];
    s += v; s2 += v * v;
  }
  __shared__ float red0[256], red1[256];
  red0[threadIdx.x] = s; red1[threadIdx.x] = s2;
  __syncthreads();
  for (int st = 128; st > 0; st >>= 1) {
    if (threadIdx.x < st) {
      red0[threadIdx.x] += red0[threadIdx.x + st];
      red1[threadIdx.x] += red1[threadIdx.x + st];
    }
    __syncthreads();
  }
  const float mean = red0[0] / (float)N;
  const float var  = fmaxf(red1[0] / (float)N - mean * mean, 0.f);
  const float rstd = rsqrtf(var + EPSV);
  for (int i = threadIdx.x; i < N; i += 256) {
    const int c  = g * CPG + i / HWN;
    const int hw = i % HWN;
    const float v = (xp[i] - mean) * rstd * gw[c] + gb[c];
    h16[((size_t)b * HWN + hw) * CC + c] = (_Float16)v;
  }
}

// ---------------- Kernel 2: fused QKV projection ----------------
// A tile async-staged in LDS (shared by 4 waves); weight B-fragments
// register double-buffered so loads run one WMMA-triplet ahead.
__global__ __launch_bounds__(128) void qkv_kernel(const _Float16* __restrict__ h16,
    const _Float16* __restrict__ wq, const float* __restrict__ qb,
    const _Float16* __restrict__ wk, const float* __restrict__ kb,
    const _Float16* __restrict__ wv, const float* __restrict__ vb,
    _Float16* __restrict__ q16, _Float16* __restrict__ k16,
    _Float16* __restrict__ vt16) {
  __shared__ _Float16 ldsA[16 * CC];   // 16 KB
  const int tid  = threadIdx.x;
  const int w    = tid >> 5;
  const int lane = tid & 31;
  const int half = lane >> 4, r = lane & 15;
  const int m0 = blockIdx.x * 16;
  const int n0 = (blockIdx.y * 4 + w) * 16;
  const int b  = blockIdx.z;

  const _Float16* A = h16 + ((size_t)b * HWN + m0) * CC;
  for (int j = 0; j < 8; ++j) {       // 16KB via async engine
    const int c = tid + j * 128;
    async_b128(A + c * 8, ldsA + c * 8);
  }
  WAIT_ASYNC_0();
  __syncthreads();

  const _Float16* Bq = wq + (size_t)(n0 + r) * CC;
  const _Float16* Bk = wk + (size_t)(n0 + r) * CC;
  const _Float16* Bv = wv + (size_t)(n0 + r) * CC;
  const _Float16* Ar = ldsA + (size_t)r * CC;

  v8f dq = {}, dk = {}, dv = {};
  int ko = half * 8;
  v16h a  = ldfrag(Ar + ko);
  v16h bq = ldfrag(Bq + ko);
  v16h bk2 = ldfrag(Bk + ko);
  v16h bv2 = ldfrag(Bv + ko);
  for (int kk = 0; kk < CC / 32 - 1; ++kk) {
    const int kn = (kk + 1) * 32 + half * 8;
    const v16h a_n  = ldfrag(Ar + kn);
    const v16h bq_n = ldfrag(Bq + kn);
    const v16h bk_n = ldfrag(Bk + kn);
    const v16h bv_n = ldfrag(Bv + kn);
    dq = wmma16(a, bq, dq);
    dk = wmma16(a, bk2, dk);
    dv = wmma16(a, bv2, dv);
    a = a_n; bq = bq_n; bk2 = bk_n; bv2 = bv_n;
  }
  dq = wmma16(a, bq, dq);
  dk = wmma16(a, bk2, dk);
  dv = wmma16(a, bv2, dv);

  const int   n      = n0 + r;
  const float qscale = 0.044194173824159216f; // 1/sqrt(512)
  const float qbias = qb[n], kbias = kb[n], vbias = vb[n];
  for (int s = 0; s < 8; ++s) {
    const int m = m0 + s + half * 8;
    q16[((size_t)b * HWN + m) * CC + n]  = (_Float16)((dq[s] + qbias) * qscale);
    k16[((size_t)b * HWN + m) * CC + n]  = (_Float16)(dk[s] + kbias);
    vt16[((size_t)b * CC + n) * HWN + m] = (_Float16)(dv[s] + vbias);
  }
}

// ---------------- Kernel 3: flash attention ----------------
// 4-wave WG per 16 queries; wave w owns channels [w*128, w*128+128).
// K/V tiles double-buffered in LDS via global_load_async_to_lds_b128.
__device__ __forceinline__ void stage_tile(const _Float16* Kg, const _Float16* Vg,
                                           int kb_, _Float16* kbuf, _Float16* vbuf,
                                           int tid) {
  // K tile: 32 keys x 512 ch = 32KB, contiguous in global -> linear copy
  const _Float16* ksrc = Kg + (size_t)kb_ * CC;
  for (int j = 0; j < 16; ++j) {
    const int c = tid + j * 128;        // 8-half chunk index
    async_b128(ksrc + c * 8, kbuf + c * 8);
  }
  // V tile: 512 ch rows x 32 keys (Vt row-major), 4 rows/thread
  for (int rr = 0; rr < 4; ++rr) {
    const int ch = tid * 4 + rr;
    const _Float16* vsrc = Vg + (size_t)ch * HWN + kb_;
    _Float16* vdst = vbuf + ch * 32;
    for (int c2 = 0; c2 < 4; ++c2)
      async_b128(vsrc + c2 * 8, vdst + c2 * 8);
  }
}

__global__ __launch_bounds__(128) void attn_kernel(const _Float16* __restrict__ q16,
                                                   const _Float16* __restrict__ k16,
                                                   const _Float16* __restrict__ vt16,
                                                   _Float16* __restrict__ o16) {
  extern __shared__ _Float16 smem[];
  _Float16* ldsQ  = smem;                 // 16*512
  _Float16* ldsK0 = ldsQ  + 16 * CC;      // 32*512
  _Float16* ldsK1 = ldsK0 + 32 * CC;
  _Float16* ldsV0 = ldsK1 + 32 * CC;      // 512*32
  _Float16* ldsV1 = ldsV0 + CC * 32;
  _Float16* ldsPb = ldsV1 + CC * 32;      // 4 * 16*32

  const int tid  = threadIdx.x;
  const int w    = tid >> 5;
  const int lane = tid & 31;
  const int half = lane >> 4, r = lane & 15;
  const int tile = blockIdx.x;
  const int b  = tile / (HWN / 16);
  const int q0 = (tile % (HWN / 16)) * 16;

  const _Float16* Qg = q16  + ((size_t)b * HWN + q0) * CC;
  const _Float16* Kg = k16  + (size_t)b * HWN * CC;
  const _Float16* Vg = vt16 + (size_t)b * CC * HWN;

  // Prologue: async-stage Q tile, then K/V tile 0 into buffer 0
  for (int j = 0; j < 8; ++j) {
    const int c = tid + j * 128;
    async_b128(Qg + c * 8, ldsQ + c * 8);
  }
  stage_tile(Kg, Vg, 0, ldsK0, ldsV0, tid);

  const int n0w = w * 128;
  v8f acc[8] = {};
  float mrun[8], lrun[8];
  for (int s = 0; s < 8; ++s) { mrun[s] = -3.0e38f; lrun[s] = 0.f; }
  _Float16* P = ldsPb + w * (16 * 32);

  for (int it = 0; it < HWN / 32; ++it) {
    const int kb_ = it * 32;
    _Float16* kbuf = (it & 1) ? ldsK1 : ldsK0;
    _Float16* vbuf = (it & 1) ? ldsV1 : ldsV0;
    if (it + 1 < HWN / 32) {  // issue next tile, then wait for current (<=32 left)
      stage_tile(Kg, Vg, kb_ + 32, (it & 1) ? ldsK0 : ldsK1,
                 (it & 1) ? ldsV0 : ldsV1, tid);
      WAIT_ASYNC_32();
    } else {
      WAIT_ASYNC_0();
    }
    __syncthreads();  // current tile visible to all waves

    // ---- S(16x32) = Qtile x Ktile^T over K = 512 (from LDS) ----
    v8f s0 = {}, s1 = {};
    for (int kk = 0; kk < CC / 32; ++kk) {
      const int ko = kk * 32 + half * 8;
      const v16h a  = ldfrag(ldsQ + r * CC + ko);
      const v16h b0 = ldfrag(kbuf + (size_t)r * CC + ko);
      const v16h b1 = ldfrag(kbuf + (size_t)(16 + r) * CC + ko);
      s0 = wmma16(a, b0, s0);
      s1 = wmma16(a, b1, s1);
    }
    // ---- online softmax (rows live across the 16-lane half) ----
    float alpha[8];
    v8f p0, p1;
    for (int s = 0; s < 8; ++s) {
      float t = rmax16(fmaxf(s0[s], s1[s]));
      const float mnew = fmaxf(mrun[s], t);
      alpha[s] = __expf(mrun[s] - mnew);
      const float e0 = __expf(s0[s] - mnew);
      const float e1 = __expf(s1[s] - mnew);
      p0[s] = e0; p1[s] = e1;
      lrun[s] = lrun[s] * alpha[s] + rsum16(e0 + e1);
      mrun[s] = mnew;
    }
    for (int t = 0; t < 8; ++t)
      for (int s = 0; s < 8; ++s) acc[t][s] *= alpha[s];
    // ---- transpose P (D layout -> A layout) via per-wave LDS ----
    for (int s = 0; s < 8; ++s) {
      const int m = s + half * 8;
      P[m * 32 + r]      = (_Float16)p0[s];
      P[m * 32 + 16 + r] = (_Float16)p1[s];
    }
    const v16h pa = ldfrag(P + r * 32 + half * 8); // same-wave DScnt ordering
    // ---- O(16x128) += P(16x32) x V(32x128) from LDS ----
    for (int t = 0; t < 8; ++t) {
      const int ch = n0w + t * 16 + r;
      const v16h vb = ldfrag(vbuf + ch * 32 + half * 8);
      acc[t] = wmma16(pa, vb, acc[t]);
    }
    __syncthreads();  // all waves done reading before buffer is re-staged
  }
  // ---- normalize and store token-major f16 ----
  for (int t = 0; t < 8; ++t) {
    const int ch = n0w + t * 16 + r;
    for (int s = 0; s < 8; ++s) {
      const int m = q0 + s + half * 8;
      o16[((size_t)b * HWN + m) * CC + ch] = (_Float16)(acc[t][s] / lrun[s]);
    }
  }
}

// ---------------- Kernel 4: output projection + bias + residual ----------------
__global__ __launch_bounds__(128) void proj_kernel(const _Float16* __restrict__ o16,
                                                   const _Float16* __restrict__ wp,
                                                   const float* __restrict__ pb,
                                                   const float* __restrict__ x,
                                                   float* __restrict__ out) {
  __shared__ _Float16 ldsA[16 * CC];
  const int tid  = threadIdx.x;
  const int w    = tid >> 5;
  const int lane = tid & 31;
  const int half = lane >> 4, r = lane & 15;
  const int m0 = blockIdx.x * 16;
  const int n0 = (blockIdx.y * 4 + w) * 16;
  const int b  = blockIdx.z;

  const _Float16* A = o16 + ((size_t)b * HWN + m0) * CC;
  for (int j = 0; j < 8; ++j) {
    const int c = tid + j * 128;
    async_b128(A + c * 8, ldsA + c * 8);
  }
  WAIT_ASYNC_0();
  __syncthreads();

  const _Float16* Bp = wp + (size_t)(n0 + r) * CC;
  const _Float16* Ar = ldsA + (size_t)r * CC;
  v8f d = {};
  v16h a  = ldfrag(Ar + half * 8);
  v16h bb = ldfrag(Bp + half * 8);
  for (int kk = 0; kk < CC / 32 - 1; ++kk) {
    const int kn = (kk + 1) * 32 + half * 8;
    const v16h a_n  = ldfrag(Ar + kn);
    const v16h bb_n = ldfrag(Bp + kn);
    d = wmma16(a, bb, d);
    a = a_n; bb = bb_n;
  }
  d = wmma16(a, bb, d);

  const int n = n0 + r;
  const float bias = pb[n];
  const float* xr   = x   + ((size_t)b * CC + n) * HWN + m0 + half * 8;
  float*       orow = out + ((size_t)b * CC + n) * HWN + m0 + half * 8;
  for (int s = 0; s < 8; ++s) orow[s] = d[s] + bias + xr[s];
}

// ---------------- host launcher ----------------
extern "C" void kernel_launch(void* const* d_in, const int* in_sizes, int n_in,
                              void* d_out, int out_size, void* d_ws, size_t ws_size,
                              hipStream_t stream) {
  (void)in_sizes; (void)n_in; (void)out_size; (void)ws_size;
  const float* x   = (const float*)d_in[0];
  const float* gnw = (const float*)d_in[1];
  const float* gnb = (const float*)d_in[2];
  const float* qw  = (const float*)d_in[3];
  const float* qb  = (const float*)d_in[4];
  const float* kw  = (const float*)d_in[5];
  const float* kb  = (const float*)d_in[6];
  const float* vw  = (const float*)d_in[7];
  const float* vb  = (const float*)d_in[8];
  const float* pw  = (const float*)d_in[9];
  const float* pb  = (const float*)d_in[10];

  const size_t ACT = (size_t)BB * HWN * CC;
  const size_t WEL = (size_t)CC * CC;
  _Float16* h16  = (_Float16*)d_ws;
  _Float16* q16  = h16  + ACT;
  _Float16* k16  = q16  + ACT;
  _Float16* vt16 = k16  + ACT;
  _Float16* o16  = vt16 + ACT;
  _Float16* wq16 = o16  + ACT;
  _Float16* wk16 = wq16 + WEL;
  _Float16* wv16 = wk16 + WEL;
  _Float16* wp16 = wv16 + WEL;

  const int cvtGrid = (int)((WEL + 255) / 256);
  f2h_kernel<<<cvtGrid, 256, 0, stream>>>(qw, wq16, (int)WEL);
  f2h_kernel<<<cvtGrid, 256, 0, stream>>>(kw, wk16, (int)WEL);
  f2h_kernel<<<cvtGrid, 256, 0, stream>>>(vw, wv16, (int)WEL);
  f2h_kernel<<<cvtGrid, 256, 0, stream>>>(pw, wp16, (int)WEL);

  gn_kernel<<<BB * GG, 256, 0, stream>>>(x, gnw, gnb, h16);

  qkv_kernel<<<dim3(HWN / 16, CC / 64, BB), 128, 0, stream>>>(
      h16, wq16, qb, wk16, kb, wv16, vb, q16, k16, vt16);

  const size_t attn_lds = (size_t)(16 * CC + 2 * 32 * CC + 2 * CC * 32 + 4 * 16 * 32)
                          * sizeof(_Float16);  // 148 KB
  attn_kernel<<<dim3(BB * HWN / 16), 128, attn_lds, stream>>>(q16, k16, vt16, o16);

  proj_kernel<<<dim3(HWN / 16, CC / 64, BB), 128, 0, stream>>>(
      o16, wp16, pb, x, (float*)d_out);
}
